// SRULayer_15796889715352
// MI455X (gfx1250) — matrix operand order repaired and checked by reading, not compiled, not run on hardware
//
#include <hip/hip_runtime.h>
#include <hip/hip_bf16.h>
#include <math.h>

// ---------------------------------------------------------------------------
// SRU layer for MI455X (gfx1250, wave32, WMMA).
//   GEMM (M=16384,K=1024,N=3072) in f16 WMMA w/ f32 accumulate,
//   then sequential SRU scan fused with tanh/gelu, then LayerNorm.
// ---------------------------------------------------------------------------

typedef __attribute__((ext_vector_type(16))) _Float16 v16h;
typedef __attribute__((ext_vector_type(8)))  _Float16 v8h;
typedef __attribute__((ext_vector_type(4)))  _Float16 v4h;
typedef __attribute__((ext_vector_type(8)))  float    v8f;

constexpr int L    = 2048;
constexpr int Bdim = 8;
constexpr int D    = 1024;
constexpr int M    = L * Bdim;   // 16384 rows of the GEMM
constexpr int K    = D;          // 1024
constexpr int N3   = 3 * D;      // 3072

// workspace layout (bytes); total ~230 MB, g aliases U
constexpr size_t OFF_XH = 0;                                   // f16 x   (32 MB)
constexpr size_t OFF_WH = OFF_XH + (size_t)M  * K * 2;         // f16 W   (6 MB)
constexpr size_t OFF_U  = OFF_WH + (size_t)N3 * K * 2;         // f32 u   (64 MB)
constexpr size_t OFF_F  = OFF_U  + (size_t)M  * D * 4;         // f32 f   (64 MB)
constexpr size_t OFF_R  = OFF_F  + (size_t)M  * D * 4;         // f32 r   (64 MB)

// ---------------------------------------------------------------------------
// Kernel 1: f32 -> f16 conversion (vectorized, grid-stride)
// ---------------------------------------------------------------------------
__global__ __launch_bounds__(256)
void cvt_f32_to_f16(const float* __restrict__ in, _Float16* __restrict__ out, int n4)
{
    int i      = blockIdx.x * blockDim.x + threadIdx.x;
    int stride = gridDim.x * blockDim.x;
    const float4* in4 = (const float4*)in;
    v4h*          o4  = (v4h*)out;
    for (; i < n4; i += stride) {
        float4 v = in4[i];
        v4h h;
        h.x = (_Float16)v.x; h.y = (_Float16)v.y;
        h.z = (_Float16)v.z; h.w = (_Float16)v.w;
        o4[i] = h;
    }
}

// ---------------------------------------------------------------------------
// Kernel 2: WMMA GEMM.  C[m][n] = sum_k A[m][k] * W[n][k]
//   block = 256 threads (8 waves), block tile 128(M) x 128(N)
//   wave  = 64(M) x 32(N) = 4x2 subtiles of 16x16, K-loop step 32.
//   K loop fully unrolled (32 tiles): all loads are immediate-offset
//   global_load_b128 from 6 fixed base pointers, double-buffered so each
//   tile's loads are in flight while the previous tile's 8 WMMAs execute.
//   __launch_bounds__(256, 2): target 2 waves/SIMD so the scheduler keeps
//   both fragment buffers live instead of collapsing them (seen as
//   load -> s_wait_loadcnt 0x0 -> wmma with the default budget).
//   Epilogue: n<1024 -> U ; 1024<=n<2048 -> sigmoid -> F ; else sigmoid -> R
// ---------------------------------------------------------------------------

// A fragment, 16x32 f16 (contiguous K, matches B):
//   lane group 0 (lane<16): K = [k+0..15]
//   lane group 1          : K = [k+16..31]
__device__ __forceinline__ void load_a_frags(v16h a[4],
                                             const _Float16* const ap[4],
                                             int k0)
{
#pragma unroll
    for (int mi = 0; mi < 4; ++mi) {
        const _Float16* p = ap[mi] + k0;
        v8h lo = *(const v8h*)p;
        v8h hi = *(const v8h*)(p + 8);
        a[mi] = __builtin_shufflevector(lo, hi,
                0, 1, 2, 3, 4, 5, 6, 7, 8, 9, 10, 11, 12, 13, 14, 15);
    }
}

// B fragment, 32x16 f16: lane = column n, 16 contiguous K values.
// B^T[k][n] = W[n][k] -> one contiguous 32B read of W row n.
__device__ __forceinline__ void load_b_frags(v16h b[2],
                                             const _Float16* const bp[2],
                                             int k0)
{
#pragma unroll
    for (int ni = 0; ni < 2; ++ni)
        b[ni] = *(const v16h*)(bp[ni] + k0);
}

__device__ __forceinline__ void mma_tile(v8f acc[4][2], const v16h a[4],
                                         const v16h b[2])
{
#pragma unroll
    for (int mi = 0; mi < 4; ++mi)
#pragma unroll
        for (int ni = 0; ni < 2; ++ni)
            acc[mi][ni] = __builtin_amdgcn_wmma_f32_16x16x32_f16(
                /*neg_a=*/false, a[mi], /*neg_b=*/false, b[ni],
                /*c_mod=*/(short)0, acc[mi][ni],
                /*reuse_a=*/false, /*reuse_b=*/false);
}

__global__ __launch_bounds__(256, 2)
void sru_gemm_wmma(const _Float16* __restrict__ A,   // M  x K row-major (f16)
                   const _Float16* __restrict__ Bw,  // N3 x K row-major (f16)
                   float* __restrict__ U,
                   float* __restrict__ F,
                   float* __restrict__ R)
{
    const int lane = threadIdx.x & 31;
    const int wid  = threadIdx.x >> 5;
    const int sel  = lane >> 4;     // lane group: 0 or 1
    const int lrow = lane & 15;

    const int m_base = blockIdx.y * 128 + (wid & 1) * 64;
    const int n_base = blockIdx.x * 128 + (wid >> 1) * 32;

    // per-lane fragment base pointers (k-invariant part of the address);
    // with the fully unrolled K loop every load below is base + constant.
    const _Float16* ap[4];
    const _Float16* bp[2];
#pragma unroll
    for (int mi = 0; mi < 4; ++mi)
        ap[mi] = A + (size_t)(m_base + mi * 16 + lrow) * K + sel * 16;
#pragma unroll
    for (int ni = 0; ni < 2; ++ni)
        bp[ni] = Bw + (size_t)(n_base + ni * 16 + lrow) * K + sel * 16;

    v8f acc[4][2];
#pragma unroll
    for (int mi = 0; mi < 4; ++mi)
#pragma unroll
        for (int ni = 0; ni < 2; ++ni)
            acc[mi][ni] = (v8f){0.f, 0.f, 0.f, 0.f, 0.f, 0.f, 0.f, 0.f};

    // ---- fully unrolled, double-buffered K loop over 32 tiles of K=32 ----
    v16h abuf[2][4];
    v16h bbuf[2][2];
    load_a_frags(abuf[0], ap, 0);
    load_b_frags(bbuf[0], bp, 0);

#pragma unroll
    for (int t = 0; t < K / 32; ++t) {
        const int cur = t & 1;
        const int nxt = cur ^ 1;
        if (t + 1 < K / 32) {
            load_a_frags(abuf[nxt], ap, (t + 1) * 32);
            load_b_frags(bbuf[nxt], bp, (t + 1) * 32);
        }
        mma_tile(acc, abuf[cur], bbuf[cur]);   // overlaps next tile's loads
    }

    // ---- epilogue: region is uniform per block (128 | 1024) ----
    const int   region = n_base / D;                  // 0:u 1:f 2:r
    float*      dst    = (region == 0) ? U : (region == 1 ? F : R);
    const bool  sig    = (region != 0);
    const int   ncol0  = n_base - region * D;

#pragma unroll
    for (int mi = 0; mi < 4; ++mi) {
#pragma unroll
        for (int ni = 0; ni < 2; ++ni) {
            const int col = ncol0 + ni * 16 + lrow;
#pragma unroll
            for (int j = 0; j < 8; ++j) {
                // C layout: vgpr j -> row (j + 8*sel), lane%16 -> col
                const int m = m_base + mi * 16 + sel * 8 + j;
                float v = acc[mi][ni][j];
                if (sig) v = 1.0f / (1.0f + expf(-v));
                dst[(size_t)m * D + col] = v;
            }
        }
    }
}

// ---------------------------------------------------------------------------
// Kernel 3: sequential SRU recurrence + tanh/gelu, one chain per thread.
//   flat chain index i = b*D + d  (8192 chains); element idx(t) = t*8192 + i.
//   Chunked by 8 timesteps for memory-level parallelism.
//   G aliases U (same-index read-then-write per thread) -> no restrict there.
// ---------------------------------------------------------------------------
__global__ __launch_bounds__(256)
void sru_recurrence(const float* __restrict__ x,
                    const float* __restrict__ c0,
                    const float* Uin,
                    const float* __restrict__ F,
                    const float* __restrict__ R,
                    float* G,                 // may alias Uin
                    float* __restrict__ lastc)
{
    const int i = blockIdx.x * blockDim.x + threadIdx.x;   // 0..8191
    const int S = Bdim * D;                                 // 8192
    float c = c0[i];

    for (int t0 = 0; t0 < L; t0 += 8) {
        float f[8], u[8], r[8], xv[8], g[8];
#pragma unroll
        for (int j = 0; j < 8; ++j) {
            const size_t idx = (size_t)(t0 + j) * S + i;
            f[j]  = F[idx];
            u[j]  = Uin[idx];
            r[j]  = R[idx];
            xv[j] = x[idx];
        }
#pragma unroll
        for (int j = 0; j < 8; ++j) {
            c = f[j] * c + (1.0f - f[j]) * u[j];
            float hs = r[j] * tanhf(c) + (1.0f - r[j]) * xv[j];
            // exact gelu: 0.5*x*(1+erf(x/sqrt(2)))
            g[j] = 0.5f * hs * (1.0f + erff(hs * 0.70710678118654752f));
        }
#pragma unroll
        for (int j = 0; j < 8; ++j)
            G[(size_t)(t0 + j) * S + i] = g[j];
    }
    lastc[i] = c;
}

// ---------------------------------------------------------------------------
// Kernel 4: LayerNorm over D=1024 per (l,b) row. 256 threads, 4 elems each.
// ---------------------------------------------------------------------------
__global__ __launch_bounds__(256)
void sru_layernorm(const float* __restrict__ G,
                   const float* __restrict__ ln_w,
                   const float* __restrict__ ln_b,
                   float* __restrict__ out)
{
    const int row = blockIdx.x;            // 0..16383
    const int tid = threadIdx.x;

    const float4* gp = (const float4*)(G + (size_t)row * D);
    float4 v = gp[tid];

    __shared__ float red1[256];
    __shared__ float red2[256];
    red1[tid] = v.x + v.y + v.z + v.w;
    red2[tid] = v.x * v.x + v.y * v.y + v.z * v.z + v.w * v.w;
    __syncthreads();
#pragma unroll
    for (int off = 128; off > 0; off >>= 1) {
        if (tid < off) {
            red1[tid] += red1[tid + off];
            red2[tid] += red2[tid + off];
        }
        __syncthreads();
    }
    const float mu  = red1[0] * (1.0f / D);
    const float var = red2[0] * (1.0f / D) - mu * mu;
    const float inv = rsqrtf(var + 1e-5f);

    const float4 w4 = ((const float4*)ln_w)[tid];
    const float4 b4 = ((const float4*)ln_b)[tid];
    float4 o;
    o.x = (v.x - mu) * inv * w4.x + b4.x;
    o.y = (v.y - mu) * inv * w4.y + b4.y;
    o.z = (v.z - mu) * inv * w4.z + b4.z;
    o.w = (v.w - mu) * inv * w4.w + b4.w;
    ((float4*)(out + (size_t)row * D))[tid] = o;
}

// ---------------------------------------------------------------------------
extern "C" void kernel_launch(void* const* d_in, const int* in_sizes, int n_in,
                              void* d_out, int out_size, void* d_ws, size_t ws_size,
                              hipStream_t stream)
{
    (void)in_sizes; (void)n_in; (void)out_size; (void)ws_size;

    const float* x    = (const float*)d_in[0];  // (L,B,D)
    const float* c0   = (const float*)d_in[1];  // (B,D)
    const float* W    = (const float*)d_in[2];  // (3D,D)
    const float* ln_w = (const float*)d_in[3];  // (D,)
    const float* ln_b = (const float*)d_in[4];  // (D,)

    float* out   = (float*)d_out;                       // (L,B,D) flat
    float* lastc = out + (size_t)L * Bdim * D;          // then (B,D)

    char*      ws = (char*)d_ws;
    _Float16*  xh = (_Float16*)(ws + OFF_XH);
    _Float16*  wh = (_Float16*)(ws + OFF_WH);
    float*     U  = (float*)(ws + OFF_U);
    float*     F  = (float*)(ws + OFF_F);
    float*     R  = (float*)(ws + OFF_R);

    // 1) f32 -> f16 conversions
    cvt_f32_to_f16<<<4096, 256, 0, stream>>>(x, xh, (M * K) / 4);
    cvt_f32_to_f16<<<1024, 256, 0, stream>>>(W, wh, (N3 * K) / 4);

    // 2) WMMA GEMM with fused sigmoid on f/r thirds
    dim3 grid(N3 / 128, M / 128);   // (24, 128)
    sru_gemm_wmma<<<grid, 256, 0, stream>>>(xh, wh, U, F, R);

    // 3) recurrence + tanh/gelu (g overwrites U in place)
    sru_recurrence<<<(Bdim * D) / 256, 256, 0, stream>>>(x, c0, U, F, R, U, lastc);

    // 4) LayerNorm
    sru_layernorm<<<M, 256, 0, stream>>>(U, ln_w, ln_b, out);
}
